// FrameTransformer_57019985822129
// MI455X (gfx1250) — matrix operand order repaired and checked by reading, not compiled, not run on hardware
//
#include <hip/hip_runtime.h>
#include <hip/hip_bf16.h>
#include <math.h>

// ---------------------------------------------------------------------------
// FrameTransformer forward for MI455X (gfx1250, wave32, WMMA).
// GEMMs: v_wmma_f32_16x16x32_f16, double-buffered LDS fed by
// global_load_async_to_lds_b128 (ASYNCcnt pipeline, no VGPR round trip).
// Transposes: 32x32 LDS tiles, coalesced on both global sides.
// ---------------------------------------------------------------------------

typedef __attribute__((ext_vector_type(16))) _Float16 v16h;
typedef __attribute__((ext_vector_type(8)))  _Float16 v8h;
typedef __attribute__((ext_vector_type(8)))  float    v8f;

#define BZ     4
#define CIN    8
#define NBINS  1024
#define WLEN   1024
#define NHEAD  4
#define DHEAD  256
#define FF2N   4096
#define FFHN   1024
#define MTOK   (BZ * WLEN)   // 4096 token rows

enum { EP_NONE = 0, EP_LRELU = 1, EP_SILU = 2, EP_TRANS = 3, EP_ADDSCALE = 4 };

// Issue an async 16B global->LDS copy (per-lane LDS dst, per-lane global src).
__device__ __forceinline__ void async_cp16(void* lds_dst, const _Float16* gsrc) {
  const unsigned l = (unsigned)(unsigned long long)lds_dst;  // low 32b = LDS offset
  asm volatile("global_load_async_to_lds_b128 %0, %1, off"
               :: "v"(l), "v"(gsrc) : "memory");
}
#define WAIT_ASYNC(n) asm volatile("s_wait_asynccnt " #n ::: "memory")

struct GemmP {
  const _Float16* A; int lda; long sAo; long sAi; int Ain;
  const _Float16* B; int ldb; long sBo; long sBi; int Bin;   // B always [N,K]
  const float* bias;
  float* C32; _Float16* C16; int ldc; long sCo; long sCi; int Cin;
  int M, N, K;
  int ep; float scale;
};

// Batched WMMA GEMM: D = A[M,K] x B[N,K]^T (+bias/epilogue).
// 256 thr = 8 waves; block tile 128x64; wave tile 32x32 (4 WMMAs / 32-K step).
// M % 128 == 0, N % 64 == 0, K % 32 == 0 at every call site -> no bounds checks.
__launch_bounds__(256)
__global__ void k_gemm_wmma(GemmP p) {
  __shared__ _Float16 As[2][128][40];   // +8 f16 pad vs bank conflicts
  __shared__ _Float16 Bs[2][64][40];

  const int tid  = threadIdx.x;
  const int lane = tid & 31;
  const int wv   = tid >> 5;
  const int wm   = wv & 3;          // 4 waves along M (32 rows each)
  const int wn   = wv >> 2;         // 2 waves along N (32 cols each)
  const int z    = blockIdx.z;

  const _Float16* Ag = p.A + (long)(z / p.Ain) * p.sAo + (long)(z % p.Ain) * p.sAi;
  const _Float16* Bg = p.B + (long)(z / p.Bin) * p.sBo + (long)(z % p.Bin) * p.sBi;
  const long cz = (long)(z / p.Cin) * p.sCo + (long)(z % p.Cin) * p.sCi;
  float*     C32 = p.C32 ? p.C32 + cz : nullptr;
  _Float16*  C16 = p.C16 ? p.C16 + cz : nullptr;

  const int m0 = blockIdx.y * 128;
  const int n0 = blockIdx.x * 64;

  // staging map: A 128x32 (2 x b128/thread), B 64x32 (1 x b128/thread)
  const int aR = tid >> 1, aC = (tid & 1) * 16;
  const int bR = tid >> 2, bC = (tid * 8) & 31;
  const _Float16* gA = Ag + (long)(m0 + aR) * p.lda + aC;
  const _Float16* gB = Bg + (long)(n0 + bR) * p.ldb + bC;

  const int fm = lane & 15, kh = lane >> 4;
  v8f acc[2][2] = {};

  auto stage = [&](int bb, int k0) {
    async_cp16(&As[bb][aR][aC],     gA + k0);
    async_cp16(&As[bb][aR][aC + 8], gA + k0 + 8);
    async_cp16(&Bs[bb][bR][bC],     gB + k0);
  };
  auto compute = [&](int bb) {
    union { v16h v; v8h h[2]; } fa0, fa1, fb0, fb1;
    // A frag 16x32: lane m=fm, e<8 -> k=e+8*kh ; e>=8 -> k=e+8+8*kh
    const int ar0 = wm * 32 + fm, ar1 = ar0 + 16;
    fa0.h[0] = *(const v8h*)&As[bb][ar0][8 * kh];
    fa0.h[1] = *(const v8h*)&As[bb][ar0][16 + 8 * kh];
    fa1.h[0] = *(const v8h*)&As[bb][ar1][8 * kh];
    fa1.h[1] = *(const v8h*)&As[bb][ar1][16 + 8 * kh];
    // B frag 32x16: lane n=fm, e<8 -> k=e+8*kh ; e>=8 -> k=e+8+8*kh
    const int br0 = wn * 32 + fm, br1 = br0 + 16;
    fb0.h[0] = *(const v8h*)&Bs[bb][br0][8 * kh];
    fb0.h[1] = *(const v8h*)&Bs[bb][br0][16 + 8 * kh];
    fb1.h[0] = *(const v8h*)&Bs[bb][br1][8 * kh];
    fb1.h[1] = *(const v8h*)&Bs[bb][br1][16 + 8 * kh];
    acc[0][0] = __builtin_amdgcn_wmma_f32_16x16x32_f16(false, fa0.v, false, fb0.v,
                                                       (short)0, acc[0][0], false, false);
    acc[0][1] = __builtin_amdgcn_wmma_f32_16x16x32_f16(false, fa0.v, false, fb1.v,
                                                       (short)0, acc[0][1], false, false);
    acc[1][0] = __builtin_amdgcn_wmma_f32_16x16x32_f16(false, fa1.v, false, fb0.v,
                                                       (short)0, acc[1][0], false, false);
    acc[1][1] = __builtin_amdgcn_wmma_f32_16x16x32_f16(false, fa1.v, false, fb1.v,
                                                       (short)0, acc[1][1], false, false);
  };

  // double-buffered async pipeline
  stage(0, 0);
  int bb = 0;
  for (int k0 = 32; k0 < p.K; k0 += 32) {
    stage(bb ^ 1, k0);          // prefetch next tile (3 more async ops in flight)
    WAIT_ASYNC(3);              // previous tile's 3 are done (in-order completion)
    __syncthreads();
    compute(bb);
    __syncthreads();            // everyone done reading before async overwrites
    bb ^= 1;
  }
  WAIT_ASYNC(0);
  __syncthreads();
  compute(bb);

  // epilogue; C frag: element r -> m = r + 8*kh, n = fm
#pragma unroll
  for (int fi = 0; fi < 2; ++fi)
#pragma unroll
    for (int fj = 0; fj < 2; ++fj)
#pragma unroll
      for (int r = 0; r < 8; ++r) {
        const int mg = m0 + wm * 32 + fi * 16 + r + 8 * kh;
        const int ng = n0 + wn * 32 + fj * 16 + fm;
        float v = acc[fi][fj][r];
        if (p.ep == EP_TRANS) {            // C[n,m] = D[m,n]  (the "skew")
          C32[(long)ng * p.ldc + mg] = v;
          continue;
        }
        if (p.ep == EP_ADDSCALE) {         // C = (C_old + D) * scale
          v = (C32[(long)mg * p.ldc + ng] + v) * p.scale;
        } else {
          if (p.bias) v += p.bias[ng];
          if (p.ep == EP_LRELU)      v = v >= 0.f ? v : 0.01f * v;
          else if (p.ep == EP_SILU)  v = v / (1.f + expf(-v));
        }
        if (C32) C32[(long)mg * p.ldc + ng] = v;
        if (C16) C16[(long)mg * p.ldc + ng] = (_Float16)v;
      }
}

// xs[b,w,h] = sum_c x[b,c,h,w] * wv[c] ; 32x32 LDS tile transpose, both sides
// coalesced. grid = (NBINS/32, WLEN/32, B), block = 256.
__launch_bounds__(256)
__global__ void k_project(const float* __restrict__ in, const float* __restrict__ wv,
                          int Cc, float* o32, _Float16* o16) {
  __shared__ float T[32][33];
  const int b  = blockIdx.z;
  const int h0 = blockIdx.x * 32;
  const int w0 = blockIdx.y * 32;
  const int j  = threadIdx.x & 31;
  const int i0 = threadIdx.x >> 5;
  float s[4] = {0.f, 0.f, 0.f, 0.f};
  for (int c = 0; c < Cc; ++c) {
    const float* base = in + (((long)b * Cc + c) * NBINS + h0) * WLEN + w0;
    const float wc = wv[c];
#pragma unroll
    for (int t = 0; t < 4; ++t)
      s[t] += base[(long)(i0 + 8 * t) * WLEN + j] * wc;   // coalesced in w
  }
#pragma unroll
  for (int t = 0; t < 4; ++t) T[i0 + 8 * t][j] = s[t];    // T[h][w]
  __syncthreads();
#pragma unroll
  for (int t = 0; t < 4; ++t) {
    const int wi = i0 + 8 * t;
    const long o = ((long)b * WLEN + w0 + wi) * NBINS + h0 + j;  // coalesced in h
    const float v = T[j][wi];
    if (o32) o32[o] = v;
    if (o16) o16[o] = (_Float16)v;
  }
}

// depthwise conv along W on token-major [B,W,Cc]; writes f16 for next GEMM
__global__ void k_dwconv(const float* __restrict__ in, const float* __restrict__ wk,
                         int Cc, int KK, int pad, _Float16* out, long total) {
  const long t = (long)blockIdx.x * 256 + threadIdx.x;
  if (t >= total) return;
  const int  c = (int)(t % Cc);
  const long bw = t / Cc;
  const int  w = (int)(bw % WLEN);
  const int  b = (int)(bw / WLEN);
  float s = 0.f;
  for (int k = 0; k < KK; ++k) {
    const int ww = w + k - pad;
    if (ww >= 0 && ww < WLEN)
      s += in[((long)b * WLEN + ww) * Cc + c] * wk[c * KK + k];
  }
  out[t] = (_Float16)s;
}

// y = LN(a*om + r1 + r2) * g + b ; r1/r2 have their own width (zero-padded)
__launch_bounds__(256)
__global__ void k_ln(const float* a, const float* om, const float* r1, int r1C,
                     const float* r2, int r2C, const float* g, const float* bb,
                     float* o32, _Float16* o16, int Cc) {
  __shared__ float red[256];
  const long row = blockIdx.x;
  const float* ar = a + row * (long)Cc;
  float s = 0.f, s2 = 0.f;
  for (int i = threadIdx.x; i < Cc; i += 256) {
    float v = ar[i];
    if (om) v *= om[i];
    if (r1 && i < r1C) v += r1[row * (long)r1C + i];
    if (r2 && i < r2C) v += r2[row * (long)r2C + i];
    s += v; s2 += v * v;
  }
  red[threadIdx.x] = s; __syncthreads();
  for (int o = 128; o > 0; o >>= 1) {
    if (threadIdx.x < o) red[threadIdx.x] += red[threadIdx.x + o];
    __syncthreads();
  }
  const float mean = red[0] / Cc; __syncthreads();
  red[threadIdx.x] = s2; __syncthreads();
  for (int o = 128; o > 0; o >>= 1) {
    if (threadIdx.x < o) red[threadIdx.x] += red[threadIdx.x + o];
    __syncthreads();
  }
  const float var = red[0] / Cc - mean * mean;
  const float inv = rsqrtf(var + 1e-5f);
  for (int i = threadIdx.x; i < Cc; i += 256) {
    float v = ar[i];
    if (om) v *= om[i];
    if (r1 && i < r1C) v += r1[row * (long)r1C + i];
    if (r2 && i < r2C) v += r2[row * (long)r2C + i];
    const float y = (v - mean) * inv * g[i] + bb[i];
    if (o32) o32[row * (long)Cc + i] = y;
    if (o16) o16[row * (long)Cc + i] = (_Float16)y;
  }
}

__launch_bounds__(256)
__global__ void k_softmax(const float* __restrict__ S, _Float16* __restrict__ P, int Cc) {
  __shared__ float red[256];
  const long row = blockIdx.x;
  const float* sr = S + row * (long)Cc;
  _Float16*    pr = P + row * (long)Cc;
  float mx = -3.0e38f;
  for (int i = threadIdx.x; i < Cc; i += 256) mx = fmaxf(mx, sr[i]);
  red[threadIdx.x] = mx; __syncthreads();
  for (int o = 128; o > 0; o >>= 1) {
    if (threadIdx.x < o) red[threadIdx.x] = fmaxf(red[threadIdx.x], red[threadIdx.x + o]);
    __syncthreads();
  }
  mx = red[0]; __syncthreads();
  float sum = 0.f;
  for (int i = threadIdx.x; i < Cc; i += 256) sum += expf(sr[i] - mx);
  red[threadIdx.x] = sum; __syncthreads();
  for (int o = 128; o > 0; o >>= 1) {
    if (threadIdx.x < o) red[threadIdx.x] += red[threadIdx.x + o];
    __syncthreads();
  }
  const float inv = 1.f / red[0];
  for (int i = threadIdx.x; i < Cc; i += 256)
    pr[i] = (_Float16)(expf(sr[i] - mx) * inv);
}

__global__ void k_cast_f16(const float* __restrict__ in, _Float16* __restrict__ out, long n) {
  for (long i = (long)blockIdx.x * 256 + threadIdx.x; i < n; i += (long)gridDim.x * 256)
    out[i] = (_Float16)in[i];
}

// out[c*R + r] = in[r*Cc + c]  (er [DH,W] -> erT [W,DH] f16; tiny, naive OK)
__global__ void k_trans_f16(const float* __restrict__ in, _Float16* __restrict__ out,
                            int R, int Cc) {
  const long idx = (long)blockIdx.x * 256 + threadIdx.x;
  if (idx >= (long)R * Cc) return;
  const int r = (int)(idx / Cc), c = (int)(idx % Cc);
  out[(long)c * R + r] = (_Float16)in[idx];
}

// vt[b,h,d,k] = V[(b*W + k)*NBINS + h*DH + d] ; 32x32 LDS tile transpose.
// grid = (WLEN/32, DHEAD/32, B*NHEAD), block = 256.
__launch_bounds__(256)
__global__ void k_transV(const _Float16* __restrict__ in, _Float16* __restrict__ out) {
  __shared__ _Float16 T[32][34];
  const int zh = blockIdx.z;
  const int b = zh / NHEAD, h = zh % NHEAD;
  const int k0 = blockIdx.x * 32;
  const int d0 = blockIdx.y * 32;
  const int j  = threadIdx.x & 31;
  const int i0 = threadIdx.x >> 5;
#pragma unroll
  for (int t = 0; t < 4; ++t) {
    const int ki = i0 + 8 * t;
    T[ki][j] = in[((long)b * WLEN + k0 + ki) * NBINS + h * DHEAD + d0 + j]; // coalesced in d
  }
  __syncthreads();
#pragma unroll
  for (int t = 0; t < 4; ++t) {
    const int di = i0 + 8 * t;
    out[(((long)b * NHEAD + h) * DHEAD + d0 + di) * WLEN + k0 + j] = T[j][di]; // coalesced in k
  }
}

// d_out[b,0,h,w] = in[(b*W + w)*NBINS + h] ; 32x32 LDS tile transpose.
// grid = (NBINS/32, WLEN/32, B), block = 256.
__launch_bounds__(256)
__global__ void k_outT(const float* __restrict__ in, float* __restrict__ out) {
  __shared__ float T[32][33];
  const int b  = blockIdx.z;
  const int h0 = blockIdx.x * 32;
  const int w0 = blockIdx.y * 32;
  const int j  = threadIdx.x & 31;
  const int i0 = threadIdx.x >> 5;
#pragma unroll
  for (int t = 0; t < 4; ++t) {
    const int wi = i0 + 8 * t;
    T[wi][j] = in[((long)b * WLEN + w0 + wi) * NBINS + h0 + j];    // coalesced in h
  }
  __syncthreads();
#pragma unroll
  for (int t = 0; t < 4; ++t) {
    const int hi = i0 + 8 * t;
    out[((long)b * NBINS + h0 + hi) * WLEN + w0 + j] = T[j][hi];   // coalesced in w
  }
}

// ---------------------------------------------------------------------------
extern "C" void kernel_launch(void* const* d_in, const int* in_sizes, int n_in,
                              void* d_out, int out_size, void* d_ws, size_t ws_size,
                              hipStream_t stream) {
  const float* x      = (const float*)d_in[0];
  const float* mem    = (const float*)d_in[1];
  const float* in_w   = (const float*)d_in[2];
  const float* skip_w = (const float*)d_in[3];
  const float* attn_w = (const float*)d_in[4];
  const float* attn_b = (const float*)d_in[5];
  const float* er     = (const float*)d_in[6];
  const float* omega  = (const float*)d_in[7];
  const float* ln_g   = (const float*)d_in[8];
  const float* ln_b   = (const float*)d_in[9];
  const float* ln2_g  = (const float*)d_in[10];
  const float* ln2_b  = (const float*)d_in[11];
  const float* c1L_dw = (const float*)d_in[12];
  const float* c1L_pw = (const float*)d_in[13];
  const float* c1R_dw = (const float*)d_in[14];
  const float* c1R_pw = (const float*)d_in[15];
  const float* c2_dw  = (const float*)d_in[16];
  const float* c2_pw  = (const float*)d_in[17];
  const float* c3_w   = (const float*)d_in[18];
  const float* c4_w   = (const float*)d_in[19];

  // -------- workspace carve (~440 MB, 256B-aligned offsets) --------
  char* wsb = (char*)d_ws;
  size_t off = 0;
  auto carve = [&](size_t bytes) {
    size_t r = off; off += (bytes + 255) & ~(size_t)255; return r;
  };
  float*    xs32 = (float*)   (wsb + carve((size_t)MTOK * NBINS * 4));
  _Float16* xs16 = (_Float16*)(wsb + carve((size_t)MTOK * NBINS * 2));
  _Float16* ms16 = (_Float16*)(wsb + carve((size_t)MTOK * NBINS * 2));
  _Float16* q16  = (_Float16*)(wsb + carve((size_t)MTOK * NBINS * 2));
  _Float16* kk16 = (_Float16*)(wsb + carve((size_t)MTOK * NBINS * 2));
  _Float16* v16  = (_Float16*)(wsb + carve((size_t)MTOK * NBINS * 2));
  _Float16* vt16 = (_Float16*)(wsb + carve((size_t)MTOK * NBINS * 2));
  _Float16* ert16= (_Float16*)(wsb + carve((size_t)WLEN * DHEAD * 2));
  float*    S32  = (float*)   (wsb + carve((size_t)BZ * NHEAD * WLEN * WLEN * 4));
  _Float16* P16  = (_Float16*)(wsb + carve((size_t)BZ * NHEAD * WLEN * WLEN * 2));
  _Float16* o16  = (_Float16*)(wsb + carve((size_t)MTOK * NBINS * 2));
  float*    ha32 = (float*)   (wsb + carve((size_t)MTOK * NBINS * 4));
  float*    hb32 = (float*)   (wsb + carve((size_t)MTOK * NBINS * 4));
  float*    x1_32= (float*)   (wsb + carve((size_t)MTOK * NBINS * 4));
  _Float16* x1_16= (_Float16*)(wsb + carve((size_t)MTOK * NBINS * 2));
  float*    x2_32= (float*)   (wsb + carve((size_t)MTOK * NBINS * 4));
  _Float16* x2_16= (_Float16*)(wsb + carve((size_t)MTOK * NBINS * 2));
  float*    x3_32= (float*)   (wsb + carve((size_t)MTOK * NBINS * 4));
  _Float16* x3_16= (_Float16*)(wsb + carve((size_t)MTOK * NBINS * 2));
  float*    x4_32= (float*)   (wsb + carve((size_t)MTOK * NBINS * 4));
  _Float16* x4_16= (_Float16*)(wsb + carve((size_t)MTOK * NBINS * 2));
  _Float16* w16  = (_Float16*)(wsb + carve((size_t)FF2N * NBINS * 2));
  _Float16* dw16 = (_Float16*)(wsb + carve((size_t)MTOK * FF2N * 2));
  float*    hl32 = (float*)   (wsb + carve((size_t)MTOK * FF2N * 4));
  float*    hr32 = (float*)   (wsb + carve((size_t)MTOK * FFHN * 4));
  (void)ws_size; (void)in_sizes; (void)n_in; (void)out_size;

  // -------- helpers --------
  auto cast16 = [&](const float* src, _Float16* dst, long n) {
    k_cast_f16<<<dim3(2048), dim3(256), 0, stream>>>(src, dst, n);
  };
  auto gemm = [&](const _Float16* A, int lda, long sAo, long sAi, int Ain,
                  const _Float16* Bm, int ldb, long sBo, long sBi, int Bin,
                  const float* bias, float* C32, _Float16* C16, int ldc,
                  long sCo, long sCi, int Cin, int M, int N, int K, int Z,
                  int ep, float scale) {
    GemmP p{A, lda, sAo, sAi, Ain, Bm, ldb, sBo, sBi, Bin,
            bias, C32, C16, ldc, sCo, sCi, Cin, M, N, K, ep, scale};
    k_gemm_wmma<<<dim3(N / 64, M / 128, Z), dim3(256), 0, stream>>>(p);
  };

  const long sWB = (long)WLEN * NBINS;   // per-batch stride in [B,W,BINS]
  const long sWW = (long)WLEN * WLEN;    // per-(b,h) stride in [B,H,W,W]
  const dim3 TG(NBINS / 32, WLEN / 32, BZ);   // transpose grids

  // One full attention block (q from a16, k/v from kv16, result f32 -> hout)
  auto attn = [&](int ai, const _Float16* a16, const _Float16* kv16, float* hout) {
    const float* Wa = attn_w + (size_t)ai * 4 * NBINS * NBINS;
    const float* Ba = attn_b + (size_t)ai * 4 * NBINS;
    // q/k/v projections (f16 out only)
    cast16(Wa + 0 * (size_t)NBINS * NBINS, w16, (long)NBINS * NBINS);
    gemm(a16, NBINS, 0, 0, 1, w16, NBINS, 0, 0, 1, Ba + 0 * NBINS,
         nullptr, q16, NBINS, 0, 0, 1, MTOK, NBINS, NBINS, 1, EP_NONE, 1.f);
    cast16(Wa + 1 * (size_t)NBINS * NBINS, w16, (long)NBINS * NBINS);
    gemm(kv16, NBINS, 0, 0, 1, w16, NBINS, 0, 0, 1, Ba + 1 * NBINS,
         nullptr, kk16, NBINS, 0, 0, 1, MTOK, NBINS, NBINS, 1, EP_NONE, 1.f);
    cast16(Wa + 2 * (size_t)NBINS * NBINS, w16, (long)NBINS * NBINS);
    gemm(kv16, NBINS, 0, 0, 1, w16, NBINS, 0, 0, 1, Ba + 2 * NBINS,
         nullptr, v16, NBINS, 0, 0, 1, MTOK, NBINS, NBINS, 1, EP_NONE, 1.f);
    // per-head V^T for the AV GEMM ([N,K] form)
    k_transV<<<dim3(WLEN / 32, DHEAD / 32, BZ * NHEAD), 256, 0, stream>>>(v16, vt16);
    // erT : [W, DH] f16
    k_trans_f16<<<(DHEAD * WLEN + 255) / 256, 256, 0, stream>>>(
        er + (size_t)ai * DHEAD * WLEN, ert16, DHEAD, WLEN);
    // S[b,h,q,k] = (Q @ erT)[k,q]   (the pad/swap/slice "skew" == transpose)
    gemm(q16, NBINS, sWB, DHEAD, NHEAD, ert16, DHEAD, 0, 0, 1, nullptr,
         S32, nullptr, WLEN, sWW, 0, 1, WLEN, WLEN, DHEAD, BZ * NHEAD, EP_TRANS, 1.f);
    // S = (S + Q K^T) / sqrt(1024)
    gemm(q16, NBINS, sWB, DHEAD, NHEAD, kk16, NBINS, sWB, DHEAD, NHEAD, nullptr,
         S32, nullptr, WLEN, sWW, 0, 1, WLEN, WLEN, DHEAD, BZ * NHEAD,
         EP_ADDSCALE, 0.03125f);
    // softmax rows -> f16 probs
    k_softmax<<<BZ * NHEAD * WLEN, 256, 0, stream>>>(S32, P16, WLEN);
    // O[b,q,h*DH+d] = P @ V
    gemm(P16, WLEN, sWW, 0, 1,
         vt16, WLEN, (long)NHEAD * DHEAD * WLEN, (long)DHEAD * WLEN, NHEAD, nullptr,
         nullptr, o16, NBINS, sWB, DHEAD, NHEAD, WLEN, DHEAD, WLEN, BZ * NHEAD,
         EP_NONE, 1.f);
    // output projection
    cast16(Wa + 3 * (size_t)NBINS * NBINS, w16, (long)NBINS * NBINS);
    gemm(o16, NBINS, 0, 0, 1, w16, NBINS, 0, 0, 1, Ba + 3 * NBINS,
         hout, nullptr, NBINS, 0, 0, 1, MTOK, NBINS, NBINS, 1, EP_NONE, 1.f);
  };

  // -------- pipeline --------
  k_project<<<TG, 256, 0, stream>>>(x,   in_w,   CIN, xs32, xs16);
  k_project<<<TG, 256, 0, stream>>>(mem, skip_w, CIN, nullptr, ms16);

  attn(0, xs16, xs16, ha32);           // self_attn1
  attn(1, xs16, ms16, hb32);           // enc_attn1
  k_ln<<<MTOK, 256, 0, stream>>>(xs32, omega + 0 * NBINS, ha32, NBINS, hb32, NBINS,
                                 ln_g + 0 * NBINS, ln_b + 0 * NBINS,
                                 x1_32, x1_16, NBINS);

  // FF: c1L (dw11 + pw -> lrelu), c1R (dw7 + pw), LN2, c2 (dw7 + pw)
  k_dwconv<<<(MTOK * NBINS + 255) / 256, 256, 0, stream>>>(
      x1_32, c1L_dw, NBINS, 11, 5, dw16, (long)MTOK * NBINS);
  cast16(c1L_pw, w16, (long)FF2N * NBINS);
  gemm(dw16, NBINS, 0, 0, 1, w16, NBINS, 0, 0, 1, nullptr,
       hl32, nullptr, FF2N, 0, 0, 1, MTOK, FF2N, NBINS, 1, EP_LRELU, 1.f);
  k_dwconv<<<(MTOK * NBINS + 255) / 256, 256, 0, stream>>>(
      x1_32, c1R_dw, NBINS, 7, 3, dw16, (long)MTOK * NBINS);
  cast16(c1R_pw, w16, (long)FFHN * NBINS);
  gemm(dw16, NBINS, 0, 0, 1, w16, NBINS, 0, 0, 1, nullptr,
       hr32, nullptr, FFHN, 0, 0, 1, MTOK, FFHN, NBINS, 1, EP_NONE, 1.f);
  k_ln<<<MTOK, 256, 0, stream>>>(hl32, nullptr, hr32, FFHN, nullptr, 0,
                                 ln2_g, ln2_b, hl32, nullptr, FF2N);  // in place
  k_dwconv<<<(MTOK * FF2N + 255) / 256, 256, 0, stream>>>(
      hl32, c2_dw, FF2N, 7, 3, dw16, (long)MTOK * FF2N);
  cast16(c2_pw, w16, (long)NBINS * FF2N);
  gemm(dw16, FF2N, 0, 0, 1, w16, FF2N, 0, 0, 1, nullptr,
       ha32, nullptr, NBINS, 0, 0, 1, MTOK, NBINS, FF2N, 1, EP_NONE, 1.f);
  k_ln<<<MTOK, 256, 0, stream>>>(x1_32, omega + 1 * NBINS, ha32, NBINS, nullptr, 0,
                                 ln_g + 1 * NBINS, ln_b + 1 * NBINS,
                                 x2_32, x2_16, NBINS);

  attn(2, x2_16, x2_16, ha32);         // self_attn2
  k_ln<<<MTOK, 256, 0, stream>>>(x2_32, omega + 2 * NBINS, ha32, NBINS, nullptr, 0,
                                 ln_g + 2 * NBINS, ln_b + 2 * NBINS,
                                 x3_32, x3_16, NBINS);
  attn(3, x3_16, ms16, ha32);          // enc_attn2
  k_ln<<<MTOK, 256, 0, stream>>>(x3_32, omega + 3 * NBINS, ha32, NBINS, nullptr, 0,
                                 ln_g + 3 * NBINS, ln_b + 3 * NBINS,
                                 x4_32, x4_16, NBINS);

  // conv3 (silu) -> conv4 -> final LN -> transpose out
  cast16(c3_w, w16, (long)FF2N * NBINS);
  gemm(x4_16, NBINS, 0, 0, 1, w16, NBINS, 0, 0, 1, nullptr,
       nullptr, dw16, FF2N, 0, 0, 1, MTOK, FF2N, NBINS, 1, EP_SILU, 1.f);
  cast16(c4_w, w16, (long)NBINS * FF2N);
  gemm(dw16, FF2N, 0, 0, 1, w16, FF2N, 0, 0, 1, nullptr,
       ha32, nullptr, NBINS, 0, 0, 1, MTOK, NBINS, FF2N, 1, EP_NONE, 1.f);
  k_ln<<<MTOK, 256, 0, stream>>>(x4_32, omega + 4 * NBINS, ha32, NBINS, nullptr, 0,
                                 ln_g + 4 * NBINS, ln_b + 4 * NBINS,
                                 x1_32 /*reuse*/, nullptr, NBINS);
  k_outT<<<TG, 256, 0, stream>>>(x1_32, (float*)d_out);
}
